// M_GCN_t_noHGNN_4784593568244
// MI455X (gfx1250) — matrix operand order repaired and checked by reading, not compiled, not run on hardware
//
#include <hip/hip_runtime.h>

#define V_ 3
#define N_ 20000
#define F_ 256
#define H_ 256
#define E_ 160000
#define C_ 32
#define EE_ (E_ + N_)      /* edges + self loops per view */
#define M_ (V_ * N_)       /* 60000 flattened rows */

typedef __bf16 bhalf;
typedef __attribute__((ext_vector_type(8)))  __bf16 v8bf;
typedef __attribute__((ext_vector_type(16))) __bf16 v16bf;
typedef __attribute__((ext_vector_type(8)))  float  v8f;
typedef __attribute__((ext_vector_type(4)))  unsigned int u32x4;
typedef __attribute__((ext_vector_type(8)))  int i32x8;
typedef __attribute__((ext_vector_type(4)))  int i32x4;

__device__ __forceinline__ float leaky02(float x){ return x > 0.f ? x : 0.2f * x; }

__device__ __forceinline__ void atomicMaxF(float* addr, float val){
  int* ia = (int*)addr;
  int cur = *ia;
  while (__int_as_float(cur) < val){
    int prev = atomicCAS(ia, cur, __float_as_int(val));
    if (prev == cur) break;
    cur = prev;
  }
}

/* ---------------- conversion / fill kernels ---------------- */

__global__ void cvt_kernel(const float* __restrict__ s, bhalf* __restrict__ d, long n){
  long i = (long)blockIdx.x * 256 + threadIdx.x;
  if (i < n) d[i] = (bhalf)s[i];
}

/* src [batch][R][Cc] (f32) -> dst [batch][Cc][R] (bf16) */
__global__ void cvtT_kernel(const float* __restrict__ s, bhalf* __restrict__ d, int R, int Cc){
  long i = (long)blockIdx.x * 256 + threadIdx.x;
  long per = (long)R * Cc;
  if (i >= per) return;
  int b = blockIdx.z;
  int c = (int)(i / R);
  int r = (int)(i % R);
  d[b * per + (long)c * R + r] = (bhalf)s[b * per + (long)r * Cc + c];
}

__global__ void fill_kernel(float* p, float v, long n){
  long i = (long)blockIdx.x * 256 + threadIdx.x;
  if (i < n) p[i] = v;
}

/* ---------------- WMMA bf16 GEMM: C = A[M,K] x Bt[N,K]^T (+bias, act) ------------- */
/* block = 256 threads (8 waves) -> 128x16 output tile.                               */
/* B tile (16 cols x K) staged into LDS once per block via the Tensor Data Mover      */
/* (tensor_load_to_lds, TENSORcnt), reused by all 8 waves; A streamed from global.    */
/* A frag (16x32 bf16): lane r (+half*16): elems 0..15 = K(k0 + half*16 .. +15).     */
/* B frag (32x16 bf16): lane = col, elems 0..15 = K(k0 + half*16 .. +15) contiguous.  */

__global__ __launch_bounds__(256)
void wmma_gemm_kernel(const bhalf* __restrict__ A, const bhalf* __restrict__ Bt,
                      const float* __restrict__ bias, float* __restrict__ Cm,
                      int M, int Nc, int K,
                      long sA, long sB, long sC, long sBias, int act)
{
  __shared__ __attribute__((aligned(64))) bhalf smemB[16 * 1024]; /* 16 rows x K<=1024, 32 KB */

  int b = blockIdx.z;
  A  += (long)b * sA;
  Bt += (long)b * sB;
  Cm += (long)b * sC;
  const float* bp = bias ? bias + (long)b * sBias : (const float*)0;

  int wave = threadIdx.x >> 5;
  int lane = threadIdx.x & 31;
  int half = lane >> 4;
  int r    = lane & 15;
  int nBase = blockIdx.x * 16;

  /* ---- TDM: DMA Bt[nBase..nBase+15][0..K-1] -> LDS (D# per CDNA5 ISA ch.8) ---- */
  if (wave == 0){
    unsigned long long gaddr = (unsigned long long)(const void*)(Bt + (long)nBase * K);
    unsigned ldsOff = (unsigned)(unsigned long long)(const void*)smemB; /* addr[31:0] = LDS offset */
    unsigned td0 = (unsigned)K;                 /* tensor_dim0 (elements)      */
    unsigned td1 = 16u;                         /* tensor_dim1 (rows)          */
    unsigned long long st0 = (unsigned long long)K; /* tensor_dim0_stride      */
    u32x4 g0;
    g0[0] = 1u;                                             /* count=1, user D#     */
    g0[1] = ldsOff;                                         /* lds_addr             */
    g0[2] = (unsigned)(gaddr & 0xFFFFFFFFu);                /* global_addr[31:0]    */
    g0[3] = (unsigned)((gaddr >> 32) & 0x01FFFFFFu)         /* global_addr[56:32]   */
          | (2u << 30);                                     /* type = 2 ("image")   */
    i32x8 g1;
    g1[0] = (int)(1u << 16);                                /* data_size = 2 bytes  */
    g1[1] = (int)((td0 & 0xFFFFu) << 16);                   /* tensor_dim0[15:0]    */
    g1[2] = (int)((td0 >> 16) | ((td1 & 0xFFFFu) << 16));   /* dim0 hi | dim1 lo    */
    g1[3] = (int)((td1 >> 16) | ((unsigned)K << 16));       /* dim1 hi | tile_dim0  */
    g1[4] = (int)16u;                                       /* tile_dim1=16, dim2=0 */
    g1[5] = (int)(st0 & 0xFFFFFFFFu);                       /* stride0[31:0]        */
    g1[6] = (int)((st0 >> 32) & 0xFFFFu);                   /* stride0[47:32]       */
    g1[7] = 0;                                              /* stride1 = 0 (2D)     */
    i32x4 gz  = {0, 0, 0, 0};                               /* groups 2/3 unused    */
    i32x8 gz8 = {0, 0, 0, 0, 0, 0, 0, 0};                   /* extra group (6-arg)  */
    __builtin_amdgcn_tensor_load_to_lds(g0, g1, gz, gz, gz8, 0);
    __builtin_amdgcn_s_wait_tensorcnt((short)0);
  }
  __syncthreads();   /* publish LDS tile to all 8 waves (before any early-out) */

  int mBase = (blockIdx.y * 8 + wave) * 16;
  if (mBase >= M) return;

  const bhalf* arow = A + (long)(mBase + r) * K;
  const bhalf* brow = smemB + (long)r * K;   /* row r of B tile == output col nBase+r */

  v8f acc = {};
  for (int k0 = 0; k0 < K; k0 += 32){
    if (k0 + 32 < K)
      __builtin_prefetch(arow + k0 + 32, 0, 1);   /* global_prefetch_b8 */
    v16bf af = *(const v16bf*)(arow + k0 + half * 16);
    v16bf bfv = *(const v16bf*)(brow + k0 + half * 16);     /* ds_load_b128 x2 */
    acc = __builtin_amdgcn_wmma_f32_16x16x32_bf16(false, af, false, bfv,
                                                  (short)0, acc, false, false);
  }

  int n = nBase + r;
  float bv = bp ? bp[n] : 0.f;
  int m0 = mBase + half * 8;
#pragma unroll
  for (int i = 0; i < 8; ++i){
    float v = acc[i] + bv;
    if (act == 1) v = fmaxf(v, 0.f);
    else if (act == 2) v = tanhf(v);
    Cm[(long)(m0 + i) * Nc + n] = v;
  }
}

/* ---------------- GCN pieces ---------------- */

__global__ void deg_count_kernel(const int* __restrict__ edst, float* __restrict__ deg, long total){
  long i = (long)blockIdx.x * 256 + threadIdx.x;
  if (i >= total) return;
  int v = (int)(i / E_);
  atomicAdd(&deg[v * N_ + edst[i]], 1.0f);
}

/* grid (EE_, V_), block H_ : out[v,d,:] += h[v,s,:] * rsqrt(deg_s)*rsqrt(deg_d) */
__global__ void gcn_scatter_kernel(const float* __restrict__ Hm,
                                   const int* __restrict__ es, const int* __restrict__ ed,
                                   const float* __restrict__ deg, float* __restrict__ out){
  int e = blockIdx.x, v = blockIdx.y, j = threadIdx.x;
  int s, d;
  if (e < E_){ s = es[(long)v * E_ + e]; d = ed[(long)v * E_ + e]; } else { s = d = e - E_; }
  float nrm = rsqrtf(deg[v * N_ + s]) * rsqrtf(deg[v * N_ + d]);
  atomicAdd(&out[((long)v * N_ + d) * H_ + j], Hm[((long)v * N_ + s) * H_ + j] * nrm);
}

/* X[i] = relu(X[i] + bias[view, j]) over [V,N,H] */
__global__ void bias_act_kernel(float* __restrict__ X, const float* __restrict__ bias, long total){
  long i = (long)blockIdx.x * 256 + threadIdx.x;
  if (i >= total) return;
  int j = (int)(i % H_);
  int v = (int)(i / ((long)N_ * H_));
  X[i] = fmaxf(X[i] + bias[v * H_ + j], 0.f);
}

/* ---------------- transformer pieces ---------------- */

/* qkv [3*N,768] -> o [3*N,256]; one block per node, wave per head (x=dim32, y=head8) */
__global__ void attn_kernel(const float* __restrict__ qkv, float* __restrict__ o){
  int n = blockIdx.x;
  int d = threadIdx.x;    /* 0..31 */
  int h = threadIdx.y;    /* 0..7  */
  float q[3], k[3], vv[3];
#pragma unroll
  for (int s = 0; s < 3; ++s){
    long base = ((long)s * N_ + n) * 768 + h * 32 + d;
    q[s] = qkv[base]; k[s] = qkv[base + 256]; vv[s] = qkv[base + 512];
  }
  const float scale = 0.17677669529663687f;  /* 1/sqrt(32) */
  float p[3][3];
#pragma unroll
  for (int s = 0; s < 3; ++s){
#pragma unroll
    for (int t = 0; t < 3; ++t){
      float x = q[s] * k[t];
#pragma unroll
      for (int off = 16; off; off >>= 1) x += __shfl_xor(x, off);
      p[s][t] = x * scale;
    }
  }
#pragma unroll
  for (int s = 0; s < 3; ++s){
    float mx = fmaxf(p[s][0], fmaxf(p[s][1], p[s][2]));
    float e0 = expf(p[s][0] - mx), e1 = expf(p[s][1] - mx), e2 = expf(p[s][2] - mx);
    float z = e0 + e1 + e2;
    o[((long)s * N_ + n) * H_ + h * 32 + d] = (e0 * vv[0] + e1 * vv[1] + e2 * vv[2]) / z;
  }
}

/* out[row,:] = LN(A[row,:] + B[row,:]) * g + b ; block = 256 = H_ */
__global__ void ln_kernel(const float* __restrict__ A, const float* __restrict__ B,
                          const float* __restrict__ g, const float* __restrict__ be,
                          float* __restrict__ out){
  int row = blockIdx.x, j = threadIdx.x;
  float x = A[(long)row * H_ + j] + B[(long)row * H_ + j];
  __shared__ float s1[256], s2[256];
  s1[j] = x; s2[j] = x * x; __syncthreads();
  for (int o = 128; o; o >>= 1){
    if (j < o){ s1[j] += s1[j + o]; s2[j] += s2[j + o]; }
    __syncthreads();
  }
  float m = s1[0] * (1.f / H_);
  float var = s2[0] * (1.f / H_) - m * m;
  out[(long)row * H_ + j] = (x - m) * rsqrtf(var + 1e-5f) * g[j] + be[j];
}

/* ---------------- GAT pieces ---------------- */

__global__ void dot2_kernel(const float* __restrict__ Hm, const float* __restrict__ as,
                            const float* __restrict__ ad, float* __restrict__ hs,
                            float* __restrict__ hd){
  int row = blockIdx.x;            /* 0..V*N-1 */
  int v = row / N_;
  int j = threadIdx.x;
  float x = Hm[(long)row * H_ + j];
  float a = x * as[v * H_ + j];
  float b = x * ad[v * H_ + j];
  __shared__ float sa[256], sb[256];
  sa[j] = a; sb[j] = b; __syncthreads();
  for (int o = 128; o; o >>= 1){
    if (j < o){ sa[j] += sa[j + o]; sb[j] += sb[j + o]; }
    __syncthreads();
  }
  if (j == 0){ hs[row] = sa[0]; hd[row] = sb[0]; }
}

__global__ void edge_max_kernel(const float* __restrict__ hs, const float* __restrict__ hd,
                                const int* __restrict__ es, const int* __restrict__ ed,
                                float* __restrict__ m){
  long i = (long)blockIdx.x * 256 + threadIdx.x;
  if (i >= (long)V_ * EE_) return;
  int v = (int)(i / EE_), e = (int)(i % EE_);
  int s, d;
  if (e < E_){ s = es[(long)v * E_ + e]; d = ed[(long)v * E_ + e]; } else { s = d = e - E_; }
  atomicMaxF(&m[v * N_ + d], leaky02(hs[v * N_ + s] + hd[v * N_ + d]));
}

__global__ void edge_sum_kernel(const float* __restrict__ hs, const float* __restrict__ hd,
                                const int* __restrict__ es, const int* __restrict__ ed,
                                const float* __restrict__ m, float* __restrict__ z){
  long i = (long)blockIdx.x * 256 + threadIdx.x;
  if (i >= (long)V_ * EE_) return;
  int v = (int)(i / EE_), e = (int)(i % EE_);
  int s, d;
  if (e < E_){ s = es[(long)v * E_ + e]; d = ed[(long)v * E_ + e]; } else { s = d = e - E_; }
  float ev = leaky02(hs[v * N_ + s] + hd[v * N_ + d]);
  atomicAdd(&z[v * N_ + d], expf(ev - m[v * N_ + d]));
}

__global__ void gat_scatter_kernel(const float* __restrict__ Hm, const float* __restrict__ hs,
                                   const float* __restrict__ hd, const int* __restrict__ es,
                                   const int* __restrict__ ed, const float* __restrict__ m,
                                   const float* __restrict__ z, float* __restrict__ out){
  int e = blockIdx.x, v = blockIdx.y, j = threadIdx.x;
  int s, d;
  if (e < E_){ s = es[(long)v * E_ + e]; d = ed[(long)v * E_ + e]; } else { s = d = e - E_; }
  float ev = leaky02(hs[v * N_ + s] + hd[v * N_ + d]);
  float w = expf(ev - m[v * N_ + d]) / z[v * N_ + d];
  atomicAdd(&out[((long)v * N_ + d) * H_ + j], Hm[((long)v * N_ + s) * H_ + j] * w);
}

/* ---------------- view attention / blending ---------------- */

__global__ void dotq_kernel(const float* __restrict__ X, const float* __restrict__ q,
                            float* __restrict__ w){
  int row = blockIdx.x, j = threadIdx.x;   /* block 128 */
  float p = X[(long)row * 128 + j] * q[j];
#pragma unroll
  for (int o = 16; o; o >>= 1) p += __shfl_xor(p, o);
  __shared__ float s[4];
  if ((j & 31) == 0) s[j >> 5] = p;
  __syncthreads();
  if (j == 0) w[row] = s[0] + s[1] + s[2] + s[3];
}

__global__ void view_blend_kernel(const float* __restrict__ a1, const float* __restrict__ w0,
                                  const float* __restrict__ w1, float* __restrict__ X1,
                                  float* __restrict__ X2){
  int n = blockIdx.x, j = threadIdx.x;
  float s0[3], s1[3];
  float m0 = -1e30f, m1 = -1e30f;
#pragma unroll
  for (int s = 0; s < 3; ++s){
    s0[s] = w0[s * N_ + n]; s1[s] = w1[s * N_ + n];
    m0 = fmaxf(m0, s0[s]);  m1 = fmaxf(m1, s1[s]);
  }
  float z0 = 0.f, z1 = 0.f;
#pragma unroll
  for (int s = 0; s < 3; ++s){ s0[s] = expf(s0[s] - m0); z0 += s0[s];
                               s1[s] = expf(s1[s] - m1); z1 += s1[s]; }
  float o1 = 0.f, o2 = 0.f;
#pragma unroll
  for (int s = 0; s < 3; ++s){
    float a = a1[((long)s * N_ + n) * H_ + j];
    o1 += (s0[s] / z0) * a;
    o2 += (s1[s] / z1) * a;
  }
  X1[(long)n * H_ + j] = o1;
  X2[(long)n * H_ + j] = o2;
}

__global__ void init_gstat_kernel(float* g){ g[0] = -1e30f; g[1] = 0.f; }

__global__ void gmax_kernel(const float* __restrict__ rw, float* __restrict__ gs){
  __shared__ float sm[256];
  int i = blockIdx.x * 256 + threadIdx.x;
  sm[threadIdx.x] = (i < N_) ? rw[i] : -1e30f;
  __syncthreads();
  for (int o = 128; o; o >>= 1){
    if (threadIdx.x < o) sm[threadIdx.x] = fmaxf(sm[threadIdx.x], sm[threadIdx.x + o]);
    __syncthreads();
  }
  if (threadIdx.x == 0) atomicMaxF(&gs[0], sm[0]);
}

__global__ void gsum_kernel(const float* __restrict__ rw, float* __restrict__ gs){
  __shared__ float sm[256];
  float mx = gs[0];
  int i = blockIdx.x * 256 + threadIdx.x;
  sm[threadIdx.x] = (i < N_) ? expf(rw[i] - mx) : 0.f;
  __syncthreads();
  for (int o = 128; o; o >>= 1){
    if (threadIdx.x < o) sm[threadIdx.x] += sm[threadIdx.x + o];
    __syncthreads();
  }
  if (threadIdx.x == 0) atomicAdd(&gs[1], sm[0]);
}

__global__ void blend_g_kernel(const float* __restrict__ rw, const float* __restrict__ gs,
                               const float* __restrict__ X1, const float* __restrict__ X2,
                               float* __restrict__ G){
  int n = blockIdx.x, j = threadIdx.x;
  float aw = expf(rw[n] - gs[0]) / gs[1];
  long i = (long)n * H_ + j;
  G[i] = aw * X1[i] + (1.f - aw) * X2[i];
}

/* ---------------- final myGAT ---------------- */

__global__ void dot32_kernel(const float* __restrict__ Fm, const float* __restrict__ al,
                             const float* __restrict__ ar, float* __restrict__ fal,
                             float* __restrict__ far){
  int row = blockIdx.x, j = threadIdx.x;   /* block 32 = one wave */
  float x = Fm[(long)row * C_ + j];
  float a = x * al[j], b = x * ar[j];
#pragma unroll
  for (int o = 16; o; o >>= 1){ a += __shfl_xor(a, o); b += __shfl_xor(b, o); }
  if (j == 0){ fal[row] = a; far[row] = b; }
}

__global__ void mg_max_kernel(const float* __restrict__ fal, const float* __restrict__ far,
                              const int* __restrict__ es, const int* __restrict__ ed,
                              float* __restrict__ m){
  long i = (long)blockIdx.x * 256 + threadIdx.x;
  if (i >= EE_) return;
  int s = es[i], d = ed[i];
  atomicMaxF(&m[d], leaky02(fal[s] + far[d]));
}

__global__ void mg_sum_kernel(const float* __restrict__ fal, const float* __restrict__ far,
                              const int* __restrict__ es, const int* __restrict__ ed,
                              const float* __restrict__ m, float* __restrict__ z){
  long i = (long)blockIdx.x * 256 + threadIdx.x;
  if (i >= EE_) return;
  int s = es[i], d = ed[i];
  atomicAdd(&z[d], expf(leaky02(fal[s] + far[d]) - m[d]));
}

__global__ void mg_scatter_kernel(const float* __restrict__ Fm, const float* __restrict__ fal,
                                  const float* __restrict__ far, const int* __restrict__ es,
                                  const int* __restrict__ ed, const float* __restrict__ m,
                                  const float* __restrict__ z, float* __restrict__ rst){
  int e = blockIdx.x, j = threadIdx.x;   /* block 32 */
  int s = es[e], d = ed[e];
  float w = expf(leaky02(fal[s] + far[d]) - m[d]) / z[d];
  atomicAdd(&rst[(long)d * C_ + j], Fm[(long)s * C_ + j] * w);
}

__global__ void norml2_kernel(const float* __restrict__ rst, float* __restrict__ out){
  int n = blockIdx.x, j = threadIdx.x;   /* block 32 */
  float x = rst[(long)n * C_ + j];
  float s = x * x;
#pragma unroll
  for (int o = 16; o; o >>= 1) s += __shfl_xor(s, o);
  out[(long)n * C_ + j] = x / fmaxf(sqrtf(s), 1e-12f);
}

/* ================= host side ================= */

extern "C" void kernel_launch(void* const* d_in, const int* in_sizes, int n_in,
                              void* d_out, int out_size, void* d_ws, size_t ws_size,
                              hipStream_t stream) {
  (void)in_sizes; (void)n_in; (void)out_size;

  const float* x      = (const float*)d_in[0];
  const int*   e_src  = (const int*)d_in[1];
  const int*   e_dst  = (const int*)d_in[2];
  const int*   gsrc   = (const int*)d_in[3];
  const int*   gdst   = (const int*)d_in[4];
  const float* rand_w = (const float*)d_in[5];
  const float* gcn_W  = (const float*)d_in[6];
  const float* gcn_b  = (const float*)d_in[7];
  const float* gat_W  = (const float*)d_in[8];
  const float* gat_as = (const float*)d_in[9];
  const float* gat_ad = (const float*)d_in[10];
  const float* gat_b  = (const float*)d_in[11];
  const float* Wqkv   = (const float*)d_in[12];
  const float* bqkv   = (const float*)d_in[13];
  const float* Wo     = (const float*)d_in[14];
  const float* bo     = (const float*)d_in[15];
  const float* W1     = (const float*)d_in[16];
  const float* b1     = (const float*)d_in[17];
  const float* W2     = (const float*)d_in[18];
  const float* b2     = (const float*)d_in[19];
  const float* ln1g   = (const float*)d_in[20];
  const float* ln1b   = (const float*)d_in[21];
  const float* ln2g   = (const float*)d_in[22];
  const float* ln2b   = (const float*)d_in[23];
  const float* att_P  = (const float*)d_in[24];
  const float* att_Pb = (const float*)d_in[25];
  const float* att_q  = (const float*)d_in[26];
  const float* mg_fc  = (const float*)d_in[27];
  const float* mg_al  = (const float*)d_in[28];
  const float* mg_ar  = (const float*)d_in[29];
  const float* mg_res = (const float*)d_in[30];

  float* a0_out = (float*)d_out;
  float* a1_out = a0_out + (size_t)M_ * H_;
  float* logits = a1_out + (size_t)M_ * H_;
  float* g_out  = logits + (size_t)N_ * C_;

  char* ws = (char*)d_ws;
  size_t off = 0;
  auto take = [&](size_t bytes) -> char* {
    char* p = ws + off;
    off = (off + bytes + 255) & ~(size_t)255;
    return p;
  };

  bhalf* gcnWt = (bhalf*)take((size_t)V_ * H_ * F_ * 2);
  bhalf* gatWt = (bhalf*)take((size_t)V_ * H_ * H_ * 2);
  bhalf* WqkvB = (bhalf*)take((size_t)2 * 768 * H_ * 2);
  bhalf* WoB   = (bhalf*)take((size_t)2 * H_ * H_ * 2);
  bhalf* W1B   = (bhalf*)take((size_t)2 * 1024 * H_ * 2);
  bhalf* W2B   = (bhalf*)take((size_t)2 * H_ * 1024 * 2);
  bhalf* PtB   = (bhalf*)take((size_t)2 * 128 * H_ * 2);
  bhalf* fcT   = (bhalf*)take((size_t)C_ * H_ * 2);
  bhalf* resT  = (bhalf*)take((size_t)C_ * H_ * 2);
  bhalf* Abf   = (bhalf*)take((size_t)M_ * 1024 * 2);
  float* T0   = (float*)take((size_t)M_ * H_ * 4);
  float* T1   = (float*)take((size_t)M_ * 1024 * 4);
  float* T2   = (float*)take((size_t)M_ * H_ * 4);
  float* T3   = (float*)take((size_t)M_ * H_ * 4);
  float* deg  = (float*)take((size_t)V_ * N_ * 4);
  float* hsA  = (float*)take((size_t)V_ * N_ * 4);
  float* hdA  = (float*)take((size_t)V_ * N_ * 4);
  float* emA  = (float*)take((size_t)V_ * N_ * 4);
  float* ezA  = (float*)take((size_t)V_ * N_ * 4);
  float* w0A  = (float*)take((size_t)V_ * N_ * 4);
  float* w1A  = (float*)take((size_t)V_ * N_ * 4);
  float* X1   = (float*)take((size_t)N_ * H_ * 4);
  float* X2   = (float*)take((size_t)N_ * H_ * 4);
  float* Fbuf = (float*)take((size_t)N_ * C_ * 4);
  float* RST  = (float*)take((size_t)N_ * C_ * 4);
  float* falA = (float*)take((size_t)N_ * 4);
  float* farA = (float*)take((size_t)N_ * 4);
  float* m2A  = (float*)take((size_t)N_ * 4);
  float* z2A  = (float*)take((size_t)N_ * 4);
  float* gstat = (float*)take(256);
  if (off > ws_size) return;  /* workspace too small: bail deterministically */

  auto cvt = [&](const float* s, bhalf* d, long n){
    cvt_kernel<<<dim3((unsigned)((n + 255) / 256)), 256, 0, stream>>>(s, d, n);
  };
  auto cvtT = [&](const float* s, bhalf* d, int R, int Cc, int batch){
    cvtT_kernel<<<dim3((unsigned)(((long)R * Cc + 255) / 256), 1, batch), 256, 0, stream>>>(s, d, R, Cc);
  };
  auto fill = [&](float* p, float v, long n){
    fill_kernel<<<dim3((unsigned)((n + 255) / 256)), 256, 0, stream>>>(p, v, n);
  };
  auto gemm = [&](const bhalf* A, const bhalf* Bt, const float* bias, float* Cc,
                  int Mm, int Nn, int K, int batch, long sA, long sB, long sC, long sBias, int act){
    dim3 g(Nn / 16, (Mm + 127) / 128, batch);
    wmma_gemm_kernel<<<g, 256, 0, stream>>>(A, Bt, bias, Cc, Mm, Nn, K, sA, sB, sC, sBias, act);
  };

  /* ---- weight conversions (bf16, [out,in] layout) ---- */
  cvtT(gcn_W, gcnWt, F_, H_, V_);
  cvtT(gat_W, gatWt, H_, H_, V_);
  cvt(Wqkv, WqkvB, (long)2 * 768 * H_);
  cvt(Wo,   WoB,   (long)2 * H_ * H_);
  cvt(W1,   W1B,   (long)2 * 1024 * H_);
  cvt(W2,   W2B,   (long)2 * H_ * 1024);
  cvtT(att_P, PtB, H_, 128, 2);
  cvtT(mg_fc, fcT, H_, C_, 1);
  cvtT(mg_res, resT, H_, C_, 1);

  /* ---- Stage A: multi-view GCN + relu -> T2 ---- */
  fill(deg, 1.0f, (long)V_ * N_);   /* self loop */
  deg_count_kernel<<<dim3((V_ * E_ + 255) / 256), 256, 0, stream>>>(e_dst, deg, (long)V_ * E_);
  cvt(x, Abf, (long)M_ * F_);
  gemm(Abf, gcnWt, nullptr, T0, N_, H_, F_, V_,
       (long)N_ * F_, (long)H_ * F_, (long)N_ * H_, 0, 0);
  fill(T2, 0.f, (long)M_ * H_);
  gcn_scatter_kernel<<<dim3(EE_, V_), H_, 0, stream>>>(T0, e_src, e_dst, deg, T2);
  bias_act_kernel<<<dim3((unsigned)(((long)M_ * H_ + 255) / 256)), 256, 0, stream>>>(T2, gcn_b, (long)M_ * H_);

  /* ---- transformer encoder layer ---- */
  auto run_enc = [&](int l, const float* xin, float* out_p){
    cvt(xin, Abf, (long)M_ * H_);
    gemm(Abf, WqkvB + (size_t)l * 768 * H_, bqkv + l * 768, T1, M_, 768, H_, 1, 0, 0, 0, 0, 0);
    attn_kernel<<<dim3(N_), dim3(32, 8), 0, stream>>>(T1, T0);
    cvt(T0, Abf, (long)M_ * H_);
    gemm(Abf, WoB + (size_t)l * H_ * H_, bo + l * H_, T3, M_, H_, H_, 1, 0, 0, 0, 0, 0);
    ln_kernel<<<dim3(M_), 256, 0, stream>>>(xin, T3, ln1g + l * H_, ln1b + l * H_, T0);
    cvt(T0, Abf, (long)M_ * H_);
    gemm(Abf, W1B + (size_t)l * 1024 * H_, b1 + l * 1024, T1, M_, 1024, H_, 1, 0, 0, 0, 0, 1);
    cvt(T1, Abf, (long)M_ * 1024);
    gemm(Abf, W2B + (size_t)l * H_ * 1024, b2 + l * H_, T3, M_, H_, 1024, 1, 0, 0, 0, 0, 0);
    ln_kernel<<<dim3(M_), 256, 0, stream>>>(T0, T3, ln2g + l * H_, ln2b + l * H_, out_p);
  };

  /* ---- Stage B: encoder 0 -> a0 ---- */
  run_enc(0, T2, a0_out);

  /* ---- Stage C: multi-view GAT + relu -> T2 ---- */
  cvt(a0_out, Abf, (long)M_ * H_);
  gemm(Abf, gatWt, nullptr, T0, N_, H_, H_, V_,
       (long)N_ * H_, (long)H_ * H_, (long)N_ * H_, 0, 0);
  dot2_kernel<<<dim3(M_), 256, 0, stream>>>(T0, gat_as, gat_ad, hsA, hdA);
  fill(emA, -1e30f, (long)V_ * N_);
  fill(ezA, 0.f, (long)V_ * N_);
  fill(T2, 0.f, (long)M_ * H_);
  {
    unsigned eb = (unsigned)(((long)V_ * EE_ + 255) / 256);
    edge_max_kernel<<<dim3(eb), 256, 0, stream>>>(hsA, hdA, e_src, e_dst, emA);
    edge_sum_kernel<<<dim3(eb), 256, 0, stream>>>(hsA, hdA, e_src, e_dst, emA, ezA);
  }
  gat_scatter_kernel<<<dim3(EE_, V_), H_, 0, stream>>>(T0, hsA, hdA, e_src, e_dst, emA, ezA, T2);
  bias_act_kernel<<<dim3((unsigned)(((long)M_ * H_ + 255) / 256)), 256, 0, stream>>>(T2, gat_b, (long)M_ * H_);

  /* ---- Stage D: encoder 1 -> a1 ---- */
  run_enc(1, T2, a1_out);

  /* ---- Stage E: attention over views + random blend -> g ---- */
  cvt(a1_out, Abf, (long)M_ * H_);
  gemm(Abf, PtB, att_Pb, T0, M_, 128, H_, 1, 0, 0, 0, 0, 2);               /* tanh */
  dotq_kernel<<<dim3(M_), 128, 0, stream>>>(T0, att_q, w0A);
  gemm(Abf, PtB + (size_t)128 * H_, att_Pb + 128, T0, M_, 128, H_, 1, 0, 0, 0, 0, 1); /* relu */
  dotq_kernel<<<dim3(M_), 128, 0, stream>>>(T0, att_q + 128, w1A);
  view_blend_kernel<<<dim3(N_), 256, 0, stream>>>(a1_out, w0A, w1A, X1, X2);
  init_gstat_kernel<<<1, 1, 0, stream>>>(gstat);
  gmax_kernel<<<dim3((N_ + 255) / 256), 256, 0, stream>>>(rand_w, gstat);
  gsum_kernel<<<dim3((N_ + 255) / 256), 256, 0, stream>>>(rand_w, gstat);
  blend_g_kernel<<<dim3(N_), 256, 0, stream>>>(rand_w, gstat, X1, X2, g_out);

  /* ---- Stage F: myGATConv + L2 normalize -> logits ---- */
  cvt(g_out, Abf, (long)N_ * H_);
  gemm(Abf, fcT,  nullptr, Fbuf, N_, C_, H_, 1, 0, 0, 0, 0, 0);
  gemm(Abf, resT, nullptr, RST,  N_, C_, H_, 1, 0, 0, 0, 0, 0);
  dot32_kernel<<<dim3(N_), 32, 0, stream>>>(Fbuf, mg_al, mg_ar, falA, farA);
  fill(m2A, -1e30f, N_);
  fill(z2A, 0.f, N_);
  mg_max_kernel<<<dim3((EE_ + 255) / 256), 256, 0, stream>>>(falA, farA, gsrc, gdst, m2A);
  mg_sum_kernel<<<dim3((EE_ + 255) / 256), 256, 0, stream>>>(falA, farA, gsrc, gdst, m2A, z2A);
  mg_scatter_kernel<<<dim3(EE_), 32, 0, stream>>>(Fbuf, falA, farA, gsrc, gdst, m2A, z2A, RST);
  norml2_kernel<<<dim3(N_), 32, 0, stream>>>(RST, logits);
}
